// MutiHeadAttention_1202590843456
// MI455X (gfx1250) — matrix operand order repaired and checked by reading, not compiled, not run on hardware
//
#include <hip/hip_runtime.h>
#include <hip/hip_bf16.h>

// ---------------------------------------------------------------------------
// MHA forward for gfx1250 (MI455X): f16 WMMA (v_wmma_f32_16x16x32_f16),
// fp32 accumulate, flash-style online softmax, TDM (tensor_load_to_lds)
// staging of K/V tiles shared by all 8 waves of a block.
// ---------------------------------------------------------------------------

typedef __attribute__((ext_vector_type(16))) _Float16 v16h;
typedef __attribute__((ext_vector_type(8)))  _Float16 h8;
typedef __attribute__((ext_vector_type(8)))  float    v8f;
typedef __attribute__((ext_vector_type(4)))  unsigned int u32x4;
typedef __attribute__((ext_vector_type(8)))  int      i32x8;
typedef __attribute__((ext_vector_type(4)))  int      i32x4;

#define BB 4
#define SS 2048
#define DD 1024
#define HH 16
#define DKK 64
#define BSS (BB * SS)

// ---- fragment loaders (layouts per CDNA5 ISA §7.12.2) -----------

// A matrix 16x32 f16: lane m (=lane&15) holds row m.
// sel = lane>>4:  elems 0..7 = K = k0 + sel*8 + {0..7}
//                 elems 8..15 = K = k0 + 16 + sel*8 + {0..7}
__device__ __forceinline__ v16h load_a_frag(const _Float16* base, int ld,
                                            int row, int k0, int sel) {
  const _Float16* p = base + (size_t)row * ld + k0 + sel * 8;
  h8 lo = *(const h8*)p;
  h8 hi = *(const h8*)(p + 16);
  v16h a;
#pragma unroll
  for (int i = 0; i < 8; ++i) { a[i] = lo[i]; a[8 + i] = hi[i]; }
  return a;
}

// B matrix 32x16 f16 with B[k][n] = src[nrow][k0+k] (i.e. C = A @ src^T):
// lane n (=lane&15) holds column n; elems 0..7 = K = sel*8 + {0..7},
// elems 8..15 = K = 16 + sel*8 + {0..7} (same order as the A fragment).
__device__ __forceinline__ v16h load_b_frag(const _Float16* base, int ld,
                                            int nrow, int k0, int sel) {
  const _Float16* p = base + (size_t)nrow * ld + k0 + sel * 8;
  h8 lo = *(const h8*)p;
  h8 hi = *(const h8*)(p + 16);
  v16h b;
#pragma unroll
  for (int i = 0; i < 8; ++i) { b[i] = lo[i]; b[8 + i] = hi[i]; }
  return b;
}

__device__ __forceinline__ v8f zero_v8f() {
  v8f z;
#pragma unroll
  for (int i = 0; i < 8; ++i) z[i] = 0.0f;
  return z;
}

// ---- TDM 2D tile load: global (row-major, strided) -> LDS (compact) --------
// Descriptor packing per CDNA5 ISA (D# groups).
// 2D tile: `rows` x `rowlen` f16 elements, global row stride `stride` elems.
__device__ __forceinline__ void tdm_load_2d(unsigned long long gaddr,
                                            unsigned lds_byte_off,
                                            unsigned rows, unsigned rowlen,
                                            unsigned stride) {
  u32x4 g0;
  g0[0] = 1u;                                       // count=1, user mode
  g0[1] = lds_byte_off;                             // lds_addr (bytes)
  g0[2] = (unsigned)(gaddr & 0xFFFFFFFFull);        // global_addr[31:0]
  g0[3] = (unsigned)((gaddr >> 32) & 0x1FFFFFFFull) // global_addr[56:32]
          | 0x80000000u;                            // type=2 ("image")
  i32x8 g1;
  g1[0] = (int)(1u << 16);          // wg_mask=0, data_size=1 (2 bytes)
  g1[1] = (int)(rowlen << 16);      // tensor_dim0[15:0]  @ bits 63:48
  g1[2] = (int)(rows << 16);        // tensor_dim0 hi=0 | tensor_dim1[15:0]
  g1[3] = (int)(rowlen << 16);      // tensor_dim1 hi=0 | tile_dim0
  g1[4] = (int)rows;                // tile_dim1 | tile_dim2=0
  g1[5] = (int)stride;              // tensor_dim0_stride[31:0]
  g1[6] = 0;                        // stride hi | tensor_dim1_stride lo
  g1[7] = 0;
  i32x4 z4;
  z4[0] = 0; z4[1] = 0; z4[2] = 0; z4[3] = 0;       // 2D: groups 2/3 unused
  i32x8 z8;
#pragma unroll
  for (int i = 0; i < 8; ++i) z8[i] = 0;
  __builtin_amdgcn_tensor_load_to_lds(g0, g1, z4, z4, z8, 0);
}

// ---- fp32 -> f16 conversion ------------------------------------------------

__global__ __launch_bounds__(256) void cvt_f32_to_f16(const float* __restrict__ in,
                                                      _Float16* __restrict__ out,
                                                      int n) {
  int i = blockIdx.x * 256 + threadIdx.x;
  if (i < n) out[i] = (_Float16)in[i];
}

// ---- GEMM: C[M,N] = A[M,K] @ W[N,K]^T  (torch Linear) ----------------------
// 256 threads = 8 waves; each wave computes a 16x64 strip (4 WMMA tiles).

template <bool OUT_F32>
__global__ __launch_bounds__(256) void gemm_xwT(const _Float16* __restrict__ A,
                                                const _Float16* __restrict__ W,
                                                float* __restrict__ Cf,
                                                _Float16* __restrict__ Ch,
                                                int M, int N, int K) {
  const int lane = threadIdx.x & 31;
  const int wave = threadIdx.x >> 5;
  const int n0 = lane & 15;
  const int sel = lane >> 4;

  const int ngroups = N / 64;
  const int task = blockIdx.x * 8 + wave;
  const int mtile = task / ngroups;
  const int nb = (task % ngroups) * 64;
  if (mtile * 16 >= M) return;

  v8f acc[4];
#pragma unroll
  for (int t = 0; t < 4; ++t) acc[t] = zero_v8f();

  const int arow = mtile * 16 + n0;
  for (int k0 = 0; k0 < K; k0 += 32) {
    v16h a = load_a_frag(A, K, arow, k0, sel);
    v16h b0 = load_b_frag(W, K, nb + 0 * 16 + n0, k0, sel);
    v16h b1 = load_b_frag(W, K, nb + 1 * 16 + n0, k0, sel);
    v16h b2 = load_b_frag(W, K, nb + 2 * 16 + n0, k0, sel);
    v16h b3 = load_b_frag(W, K, nb + 3 * 16 + n0, k0, sel);
    acc[0] = __builtin_amdgcn_wmma_f32_16x16x32_f16(false, a, false, b0, (short)0, acc[0], false, false);
    acc[1] = __builtin_amdgcn_wmma_f32_16x16x32_f16(false, a, false, b1, (short)0, acc[1], false, false);
    acc[2] = __builtin_amdgcn_wmma_f32_16x16x32_f16(false, a, false, b2, (short)0, acc[2], false, false);
    acc[3] = __builtin_amdgcn_wmma_f32_16x16x32_f16(false, a, false, b3, (short)0, acc[3], false, false);
  }

  // C layout: VGPR r, lanes 0-15 -> row r, lanes 16-31 -> row r+8, col = lane&15
#pragma unroll
  for (int t = 0; t < 4; ++t) {
#pragma unroll
    for (int r = 0; r < 8; ++r) {
      const int row = mtile * 16 + r + sel * 8;
      const int col = nb + t * 16 + n0;
      if (OUT_F32) Cf[(size_t)row * N + col] = acc[t][r];
      else         Ch[(size_t)row * N + col] = (_Float16)acc[t][r];
    }
  }
}

// ---- Flash attention -------------------------------------------------------
// One block = 8 waves, all on the same (b, h); wave w owns q-rows
// [qg*128 + w*16, +16). K/V 32x64 tiles are TDM-DMA'd into LDS once per
// block and consumed by all 8 waves.

#define LDS_K_OFF 0u        // 32*64 f16 = 4096 B
#define LDS_V_OFF 4096u     // 32*64 f16 = 4096 B
#define LDS_P_OFF 8192u     // 8 waves * 16*32 f16 = 8192 B

__global__ __launch_bounds__(256) void flash_attn(const _Float16* __restrict__ Q,
                                                  const _Float16* __restrict__ K,
                                                  const _Float16* __restrict__ V,
                                                  _Float16* __restrict__ O) {
  __shared__ _Float16 smem[2048 + 2048 + 8 * 512];  // 16 KB, assumed LDS off 0
  _Float16* ldsK = smem;
  _Float16* ldsV = smem + 2048;

  const int lane = threadIdx.x & 31;
  const int wave = threadIdx.x >> 5;
  const int n0 = lane & 15;
  const int sel = lane >> 4;
  _Float16* psh = smem + 4096 + wave * 512;  // per-wave P staging (16x32)

  const int blk = blockIdx.x;        // B*H*(S/128) = 1024
  const int qg = blk & 15;           // S/128 = 16
  const int h  = (blk >> 4) & 15;
  const int b  = blk >> 8;

  const int hoff  = h * DKK;
  const int qrow0 = b * SS + qg * 128 + wave * 16;

  // Q A-fragments for K-steps 0 and 32 (reused over all key tiles)
  const v16h aq0 = load_a_frag(Q + hoff, DD, qrow0 + n0, 0, sel);
  const v16h aq1 = load_a_frag(Q + hoff, DD, qrow0 + n0, 32, sel);

  float mi[8], li[8];
  v8f acc[4];
#pragma unroll
  for (int r = 0; r < 8; ++r) { mi[r] = -1e30f; li[r] = 0.0f; }
#pragma unroll
  for (int t = 0; t < 4; ++t) acc[t] = zero_v8f();

  const float scale = 0.125f;  // 1/sqrt(64)

  for (int kt = 0; kt < SS / 32; ++kt) {
    const int krow0 = b * SS + kt * 32;

    // Wave 0 DMAs this block's K and V 32x64 tiles into LDS via TDM.
    if (wave == 0) {
      unsigned long long gaK =
          (unsigned long long)(uintptr_t)(K + (size_t)krow0 * DD + hoff);
      unsigned long long gaV =
          (unsigned long long)(uintptr_t)(V + (size_t)krow0 * DD + hoff);
      tdm_load_2d(gaK, LDS_K_OFF, 32u, 64u, (unsigned)DD);
      tdm_load_2d(gaV, LDS_V_OFF, 32u, 64u, (unsigned)DD);
      __builtin_amdgcn_s_wait_tensorcnt(0);
    }
    __syncthreads();

    // scores S[16 x 32] = Q_tile @ K_tile^T  (two 16x16 n-subtiles).
    // Issue all four B-fragment loads before the WMMAs so LDS latency
    // overlaps matrix-op execution.
    v16h bk00 = load_b_frag(ldsK, 64, n0, 0, sel);
    v16h bk10 = load_b_frag(ldsK, 64, 16 + n0, 0, sel);
    v16h bk01 = load_b_frag(ldsK, 64, n0, 32, sel);
    v16h bk11 = load_b_frag(ldsK, 64, 16 + n0, 32, sel);
    v8f s0 = zero_v8f(), s1 = zero_v8f();
    s0 = __builtin_amdgcn_wmma_f32_16x16x32_f16(false, aq0, false, bk00, (short)0, s0, false, false);
    s1 = __builtin_amdgcn_wmma_f32_16x16x32_f16(false, aq0, false, bk10, (short)0, s1, false, false);
    s0 = __builtin_amdgcn_wmma_f32_16x16x32_f16(false, aq1, false, bk01, (short)0, s0, false, false);
    s1 = __builtin_amdgcn_wmma_f32_16x16x32_f16(false, aq1, false, bk11, (short)0, s1, false, false);

    // V B-fragments depend only on the LDS V tile: hoist the gathers here so
    // they overlap the softmax VALU/TRANS chain below.
    v16h bv[4];
#pragma unroll
    for (int t = 0; t < 4; ++t) {
      const _Float16* vp = ldsV + (sel * 16) * 64 + t * 16 + n0;
#pragma unroll
      for (int e = 0; e < 16; ++e) bv[t][e] = vp[e * 64];
    }

    // online softmax per row; row r of VGPR r lives across 16 lanes of a half
#pragma unroll
    for (int r = 0; r < 8; ++r) {
      float x0 = s0[r] * scale;
      float x1 = s1[r] * scale;
      float mx = fmaxf(x0, x1);
#pragma unroll
      for (int off = 1; off < 16; off <<= 1)
        mx = fmaxf(mx, __shfl_xor(mx, off, 16));
      const float mnew = fmaxf(mi[r], mx);
      const float corr = __expf(mi[r] - mnew);
      const float p0 = __expf(x0 - mnew);
      const float p1 = __expf(x1 - mnew);
      float ps = p0 + p1;
#pragma unroll
      for (int off = 1; off < 16; off <<= 1)
        ps += __shfl_xor(ps, off, 16);
      li[r] = li[r] * corr + ps;
      mi[r] = mnew;
#pragma unroll
      for (int t = 0; t < 4; ++t) acc[t][r] *= corr;
      const int prow = r + sel * 8;
      psh[prow * 32 + n0]      = (_Float16)p0;
      psh[prow * 32 + 16 + n0] = (_Float16)p1;
    }

    // wave-private LDS transpose; DS ops are in-order per wave
    asm volatile("s_wait_dscnt 0" ::: "memory");

    // P as A-fragment (16x32) from LDS
    v16h pa;
    {
      const _Float16* pr = &psh[n0 * 32 + sel * 16];
      h8 lo = *(const h8*)pr;
      h8 hi = *(const h8*)(pr + 8);
#pragma unroll
      for (int i = 0; i < 8; ++i) { pa[i] = lo[i]; pa[8 + i] = hi[i]; }
    }

    // acc += P @ V_tile
#pragma unroll
    for (int t = 0; t < 4; ++t)
      acc[t] = __builtin_amdgcn_wmma_f32_16x16x32_f16(
          false, pa, false, bv[t], (short)0, acc[t], false, false);

    __syncthreads();  // protect LDS K/V tiles before next TDM overwrite
  }

  // epilogue: O = acc / l, back to [B*S, D] f16
#pragma unroll
  for (int t = 0; t < 4; ++t) {
#pragma unroll
    for (int r = 0; r < 8; ++r) {
      const int row = qrow0 + r + sel * 8;
      const int col = hoff + t * 16 + n0;
      O[(size_t)row * DD + col] = (_Float16)(acc[t][r] / li[r]);
    }
  }
}

// ---------------------------------------------------------------------------

extern "C" void kernel_launch(void* const* d_in, const int* in_sizes, int n_in,
                              void* d_out, int out_size, void* d_ws, size_t ws_size,
                              hipStream_t stream) {
  const float* x  = (const float*)d_in[0];
  const float* Wq = (const float*)d_in[1];
  const float* Wk = (const float*)d_in[2];
  const float* Wv = (const float*)d_in[3];
  const float* Wo = (const float*)d_in[4];
  float* out = (float*)d_out;

  const size_t NX = (size_t)BSS * DD;  // 8388608
  const size_t NW = (size_t)DD * DD;   // 1048576

  _Float16* ws  = (_Float16*)d_ws;
  _Float16* xb  = ws;                   // [BS, D]; dead after projections
  _Float16* wq  = xb + NX;
  _Float16* wk  = wq + NW;
  _Float16* wv  = wk + NW;
  _Float16* wo  = wv + NW;
  _Float16* Qh  = wo + NW;              // [BS, D]
  _Float16* Kh  = Qh + NX;
  _Float16* Vh  = Kh + NX;
  _Float16* ctx = xb;                   // reuse xb region for attention output

  // 1) convert inputs to f16
  cvt_f32_to_f16<<<(int)((NX + 255) / 256), 256, 0, stream>>>(x, xb, (int)NX);
  cvt_f32_to_f16<<<(int)((NW + 255) / 256), 256, 0, stream>>>(Wq, wq, (int)NW);
  cvt_f32_to_f16<<<(int)((NW + 255) / 256), 256, 0, stream>>>(Wk, wk, (int)NW);
  cvt_f32_to_f16<<<(int)((NW + 255) / 256), 256, 0, stream>>>(Wv, wv, (int)NW);
  cvt_f32_to_f16<<<(int)((NW + 255) / 256), 256, 0, stream>>>(Wo, wo, (int)NW);

  // 2) projections: Q/K/V = x @ W^T   (M=8192, N=1024, K=1024)
  const int gemm_blocks = ((BSS / 16) * (DD / 64)) / 8;  // 1024
  gemm_xwT<false><<<gemm_blocks, 256, 0, stream>>>(xb, wq, nullptr, Qh, BSS, DD, DD);
  gemm_xwT<false><<<gemm_blocks, 256, 0, stream>>>(xb, wk, nullptr, Kh, BSS, DD, DD);
  gemm_xwT<false><<<gemm_blocks, 256, 0, stream>>>(xb, wv, nullptr, Vh, BSS, DD, DD);

  // 3) flash attention: B*H*(S/128) = 1024 blocks of 8 waves
  const int attn_blocks = BB * HH * (SS / 128);
  flash_attn<<<attn_blocks, 256, 0, stream>>>(Qh, Kh, Vh, ctx);

  // 4) output projection, fp32 result straight to d_out
  gemm_xwT<true><<<gemm_blocks, 256, 0, stream>>>(ctx, wo, out, nullptr, BSS, DD, DD);
}
